// RNN_63136019251273
// MI455X (gfx1250) — compile-verified, matrix operand order checked by reading, not run on hardware
//
#include <hip/hip_runtime.h>
#include <hip/hip_bf16.h>

typedef __attribute__((ext_vector_type(16))) _Float16 v16h;
typedef __attribute__((ext_vector_type(8)))  float    v8f;

#define H_   64
#define G4H  256      // 4*H
#define TSTEPS 256
#define BATCH 4096
#define MROWS 32      // batch rows per workgroup
#define HSTR 72       // LDS halfword stride for h tiles (16B aligned reads, bank spread)
#define XSTR 40       // LDS halfword stride for x tile (80B rows; async writes 64B payload)
#define GSTR 264      // LDS float stride for gates
#define PW_STRIDE 16384   // f16 elements reserved per packed weight matrix
#define XH_OFFSET (1u << 20)  // xh fp16 tensor starts 1MB into workspace

__device__ __forceinline__ float sigf(float x) {
    return 1.0f / (1.0f + __expf(-x));
}

__device__ __forceinline__ void wait_async0() {
#if __has_builtin(__builtin_amdgcn_s_wait_asynccnt)
    __builtin_amdgcn_s_wait_asynccnt(0);
#else
    asm volatile("s_wait_asynccnt 0x0" ::: "memory");
#endif
}

// Build A operand (16x32 f16) for one lane per the CDNA5 16-bit A layout:
// lane L: row = mtile*16 + (L&15); kgroup = L>>4;
//   VGPR0-3 hold K = kc*32 + kg*8 + 0..7  (8 consecutive halves -> 16B ds load)
//   VGPR4-7 hold K = kc*32 + 16 + kg*8 + 0..7
__device__ __forceinline__ v16h load_a_tile(const _Float16* base, int stride,
                                            int mtile, int kc, int lane) {
    int row = mtile * 16 + (lane & 15);
    int kg  = lane >> 4;
    const _Float16* p = base + row * stride + kc * 32 + kg * 8;
    v16h a;
#pragma unroll
    for (int i = 0; i < 8; ++i) {
        a[i]     = p[i];
        a[8 + i] = p[16 + i];
    }
    return a;
}

// Pack W[4H x Din] (row-major f32) into per-lane contiguous fp16 B operands.
// dst index: ((nt*KC + kc)*32 + lane)*16 + j ;  value = W[nt*16 + (lane&15)][K]
// with K = kc*32 + (vj<4 ? vj*2 : 16+(vj-4)*2) + (lane>>4)*8 + (j&1); zero-pad K>=Din.
__global__ void pack_w_kernel(const float* __restrict__ W, int Din, int KC,
                              _Float16* __restrict__ dst) {
    int idx = blockIdx.x * 256 + threadIdx.x;
    int total = 16 * KC * 32 * 16;
    if (idx >= total) return;
    int j    = idx & 15;
    int lane = (idx >> 4) & 31;
    int rem  = idx >> 9;           // nt*KC + kc
    int kc   = rem % KC;
    int nt   = rem / KC;
    int n    = nt * 16 + (lane & 15);
    int kg   = lane >> 4;
    int vj   = j >> 1;
    int koff = (vj < 4 ? vj * 2 : 16 + (vj - 4) * 2) + kg * 8 + (j & 1);
    int K    = kc * 32 + koff;
    float v  = (K < Din) ? W[n * Din + K] : 0.0f;
    dst[idx] = (_Float16)v;
}

__global__ void bias_sum_kernel(const float* __restrict__ bih,
                                const float* __restrict__ bhh,
                                float* __restrict__ dst) {
    int i = threadIdx.x;
    if (i < G4H) dst[i] = bih[i] + bhh[i];
}

// x [B,T,28] f32 -> xh [B,T,32] f16, K padded with zeros (one-shot, HBM-bound).
__global__ void xpack_kernel(const float* __restrict__ x, _Float16* __restrict__ xh) {
    size_t idx = (size_t)blockIdx.x * 256 + threadIdx.x;   // over B*T*32
    int col    = (int)(idx & 31);
    size_t rbt = idx >> 5;
    float v    = (col < 28) ? x[rbt * 28 + col] : 0.0f;
    xh[idx]    = (_Float16)v;
}

// Persistent fused 4-layer LSTM. One WG = 32 batch rows for all T timesteps,
// all 4 layers. h/c state resident in LDS/registers; x tiles arrive via
// double-buffered GLOBAL_LOAD_ASYNC_TO_LDS_B64 (ASYNCcnt), weights from L2.
__global__ __launch_bounds__(256)
void lstm_persistent_kernel(const _Float16* __restrict__ xh,      // [B,T,32] f16
                            const _Float16* __restrict__ pw_ih,   // 4 packed, stride PW_STRIDE
                            const _Float16* __restrict__ pw_hh,   // 4 packed, stride PW_STRIDE
                            const float* __restrict__ bsum,       // [4][256]
                            const float* __restrict__ W_out,      // [10][64]
                            const float* __restrict__ b_out,      // [10]
                            float* __restrict__ out) {            // [B,10]
    __shared__ _Float16 sh_h[4][MROWS][HSTR];
    __shared__ _Float16 sh_x[2][MROWS][XSTR];
    __shared__ float    sh_g[MROWS][GSTR];

    const int tid    = threadIdx.x;
    const int lane   = tid & 31;
    const int wave   = tid >> 5;
    const int mtile  = wave >> 2;         // 0..1
    const int ntbase = (wave & 3) * 4;    // 4 consecutive N-tiles: one gate strip per wave
    const int b0     = blockIdx.x * MROWS;

    // elementwise mapping: each thread owns 8 consecutive hidden cols in one row
    const int erow = tid >> 3;
    const int ecb  = (tid & 7) * 8;

    // async x-staging mapping: each thread moves 8 bytes of one 64B row payload
    const int xrow = tid >> 3;
    const int xsub = (tid & 7) * 8;       // byte offset within row

    // issue the async copy of x tile for timestep t into buffer buf
    auto stage_x = [&](int t, int buf) {
        const char* gsrc = (const char*)(xh + ((size_t)(b0 + xrow) * TSTEPS + t) * 32) + xsub;
        unsigned ldsa = (unsigned)(size_t)&sh_x[buf][xrow][0] + (unsigned)xsub;
        asm volatile("global_load_async_to_lds_b64 %0, %1, off"
                     :: "v"(ldsa), "v"(gsrc) : "memory");
    };

    // zero initial hidden state
    for (int i = tid; i < 4 * MROWS * HSTR; i += 256)
        (&sh_h[0][0][0])[i] = (_Float16)0.0f;

    float creg[4][8];
#pragma unroll
    for (int l = 0; l < 4; ++l)
#pragma unroll
        for (int k = 0; k < 8; ++k) creg[l][k] = 0.0f;

    // hoist per-tile bias values (loop-invariant across all timesteps)
    float bv[4][4];
#pragma unroll
    for (int l = 0; l < 4; ++l)
#pragma unroll
        for (int i = 0; i < 4; ++i)
            bv[l][i] = bsum[l * G4H + (ntbase + i) * 16 + (lane & 15)];

    stage_x(0, 0);   // prefetch first tile

    for (int t = 0; t < TSTEPS; ++t) {
        wait_async0();        // this wave's async writes landed in LDS
        __syncthreads();      // ... and everyone else's too (also orders sh_h init at t=0)
        if (t + 1 < TSTEPS) stage_x(t + 1, (t + 1) & 1);   // overlap t+1 fetch with t compute
        const int xbuf = t & 1;

        for (int l = 0; l < 4; ++l) {
            const int KCin = (l == 0) ? 1 : 2;
            const _Float16* inbase   = (l == 0) ? &sh_x[xbuf][0][0] : &sh_h[l - 1][0][0];
            const int       instride = (l == 0) ? XSTR : HSTR;
            const _Float16* pwih     = pw_ih + l * PW_STRIDE;
            const _Float16* pwhh     = pw_hh + l * PW_STRIDE;

            // ---- accumulators seeded with bias (register splat, no loads) ----
            v8f acc[4];
#pragma unroll
            for (int i = 0; i < 4; ++i) {
#pragma unroll
                for (int r = 0; r < 8; ++r) acc[i][r] = bv[l][i];
            }

            // ---- input GEMM: in_t @ W_ih^T ----
            for (int kc = 0; kc < KCin; ++kc) {
                v16h a = load_a_tile(inbase, instride, mtile, kc, lane);
#pragma unroll
                for (int i = 0; i < 4; ++i) {
                    const v16h bm = *(const v16h*)(pwih +
                        (((size_t)(ntbase + i) * KCin + kc) * 32 + lane) * 16);
                    acc[i] = __builtin_amdgcn_wmma_f32_16x16x32_f16(
                        false, a, false, bm, (short)0, acc[i], false, false);
                }
            }

            // ---- recurrent GEMM: h_{t-1} @ W_hh^T (K = 64 -> 2 chunks) ----
#pragma unroll
            for (int kc = 0; kc < 2; ++kc) {
                v16h a = load_a_tile(&sh_h[l][0][0], HSTR, mtile, kc, lane);
#pragma unroll
                for (int i = 0; i < 4; ++i) {
                    const v16h bm = *(const v16h*)(pwhh +
                        (((size_t)(ntbase + i) * 2 + kc) * 32 + lane) * 16);
                    acc[i] = __builtin_amdgcn_wmma_f32_16x16x32_f16(
                        false, a, false, bm, (short)0, acc[i], false, false);
                }
            }

            // ---- spill raw gates to LDS (C layout: col=lane&15, rows via kg*8+r) ----
            {
                int grow = mtile * 16 + (lane >> 4) * 8;
#pragma unroll
                for (int i = 0; i < 4; ++i) {
                    int colg = (ntbase + i) * 16 + (lane & 15);
#pragma unroll
                    for (int r = 0; r < 8; ++r)
                        sh_g[grow + r][colg] = acc[i][r];
                }
            }
            __syncthreads();

            // ---- elementwise: gate activations, c/h update ----
#pragma unroll
            for (int k = 0; k < 8; ++k) {
                int j = ecb + k;
                float ig = sigf(sh_g[erow][j]);
                float fg = sigf(sh_g[erow][64 + j]);
                float gg = tanhf(sh_g[erow][128 + j]);
                float og = sigf(sh_g[erow][192 + j]);
                float c  = fg * creg[l][k] + ig * gg;
                creg[l][k] = c;
                sh_h[l][erow][j] = (_Float16)(og * tanhf(c));
            }
            __syncthreads();
        }
    }

    // ---- final linear head on h3 at t = T-1 ----
    for (int idx = tid; idx < MROWS * 10; idx += 256) {
        int row = idx / 10;
        int o   = idx - row * 10;
        float a = b_out[o];
#pragma unroll
        for (int j = 0; j < H_; ++j)
            a += (float)sh_h[3][row][j] * W_out[o * H_ + j];
        out[(size_t)(b0 + row) * 10 + o] = a;
    }
}

extern "C" void kernel_launch(void* const* d_in, const int* in_sizes, int n_in,
                              void* d_out, int out_size, void* d_ws, size_t ws_size,
                              hipStream_t stream) {
    (void)in_sizes; (void)n_in; (void)out_size; (void)ws_size;

    const float* x     = (const float*)d_in[0];
    const float* W_out = (const float*)d_in[17];
    const float* b_out = (const float*)d_in[18];

    char* ws = (char*)d_ws;
    _Float16* pw_ih = (_Float16*)ws;                               // 4 * 16384 f16
    _Float16* pw_hh = (_Float16*)(ws + 4 * PW_STRIDE * 2);         // 4 * 16384 f16
    float*    bsum  = (float*)(ws + 8 * PW_STRIDE * 2);            // 4 * 256 f32
    _Float16* xh    = (_Float16*)(ws + XH_OFFSET);                 // [B,T,32] f16 = 64MB

    for (int l = 0; l < 4; ++l) {
        const float* Wih = (const float*)d_in[1 + l * 4 + 0];
        const float* Whh = (const float*)d_in[1 + l * 4 + 1];
        const float* bih = (const float*)d_in[1 + l * 4 + 2];
        const float* bhh = (const float*)d_in[1 + l * 4 + 3];
        int Din = (l == 0) ? 28 : 64;
        int KC  = (l == 0) ? 1 : 2;
        int total = 16 * KC * 32 * 16;
        pack_w_kernel<<<(total + 255) / 256, 256, 0, stream>>>(
            Wih, Din, KC, pw_ih + (size_t)l * PW_STRIDE);
        pack_w_kernel<<<64, 256, 0, stream>>>(
            Whh, H_, 2, pw_hh + (size_t)l * PW_STRIDE);
        bias_sum_kernel<<<1, 256, 0, stream>>>(bih, bhh, bsum + l * G4H);
    }

    // one-shot fp16 conversion+pad of x (HBM-bound, ~8us at 23.3 TB/s)
    xpack_kernel<<<(BATCH * TSTEPS * 32) / 256, 256, 0, stream>>>(x, xh);

    lstm_persistent_kernel<<<BATCH / MROWS, 256, 0, stream>>>(
        xh, pw_ih, pw_hh, bsum, W_out, b_out, (float*)d_out);
}
